// VSSblock_32581621908218
// MI455X (gfx1250) — compile-verified, matrix operand order checked by reading, not run on hardware
//
#include <hip/hip_runtime.h>

// ---------------------------------------------------------------------------
// VSS (VMamba) block x2 + maxpool for MI455X (gfx1250, wave32, WMMA).
// GEMMs use v_wmma_f32_16x16x32_f16 (f32 accumulate, f16 operands staged in
// LDS). Tile staging uses GLOBAL_LOAD_ASYNC_TO_LDS_B128 (ASYNCcnt) when the
// toolchain declares the builtin, else global_load+ds_store fallback.
// Selective scan: 16-lane groups, one state n per lane, shfl-xor butterfly.
// ---------------------------------------------------------------------------

typedef __attribute__((ext_vector_type(16))) _Float16 v16h;
typedef __attribute__((ext_vector_type(8)))  _Float16 v8h;
typedef __attribute__((ext_vector_type(8)))  float    v8f;
typedef __attribute__((ext_vector_type(4)))  int      v4i;

#define D_MODEL 96
#define D_INNER 192
#define D_STATE 16
#define DT_RANK 6
#define KDIRS   4
#define BATCH   4
#define HH      64
#define WW      64
#define LSEQ    4096   // HH*WW
#define BL      16384  // BATCH*LSEQ

#define GLOBAL_AS __attribute__((address_space(1)))
#define LDS_AS    __attribute__((address_space(3)))

typedef GLOBAL_AS v4i* gv4i_p;
typedef LDS_AS    v4i* lv4i_p;

#if __has_builtin(__builtin_amdgcn_global_load_async_to_lds_b128)
#define HAVE_ASYNC_LDS 1
#else
#define HAVE_ASYNC_LDS 0
#endif

__device__ __forceinline__ void wait_async0()
{
#if __has_builtin(__builtin_amdgcn_s_wait_asynccnt)
  __builtin_amdgcn_s_wait_asynccnt(0);
#else
  asm volatile("s_wait_asynccnt 0x0" ::: "memory");
#endif
}

// ---------------------------------------------------------------------------
// Shared WMMA GEMM mainloop: Out[64x64] += A[64xK] * B[64xK]^T for this block.
// A row-major [M][K] f16, B row-major [N][K] f16 (weights are [out][in]).
// 128 threads = 4 waves; wave w owns rows w*16..w*16+15, all 64 cols.
// LDS rows padded to 40 halfs (80B) for bank spread.
// ---------------------------------------------------------------------------
__device__ __forceinline__ void gemm_mainloop(
    const _Float16* __restrict__ Abase, int lda,
    const _Float16* __restrict__ Bbase, int ldb, int Kdim,
    _Float16* As, _Float16* Bs, v8f* acc)
{
  const int tid  = threadIdx.x;
  const int row2 = tid >> 1;          // 0..63
  const int colh = (tid & 1) << 4;    // 0 or 16 (halfs)
  const int lane = tid & 31;
  const int m16  = lane & 15;
  const int sel  = lane >> 4;         // K-half select per WMMA layout
  const int wave = tid >> 5;

  for (int k0 = 0; k0 < Kdim; k0 += 32) {
    __syncthreads();                  // all waves done reading previous tile
#if HAVE_ASYNC_LDS
    // Direct global->LDS async copy (ASYNCcnt), no VGPR round-trip.
    __builtin_amdgcn_global_load_async_to_lds_b128(
        (gv4i_p)(Abase + (size_t)row2 * lda + k0 + colh),
        (lv4i_p)(As + row2 * 40 + colh), 0, 0);
    __builtin_amdgcn_global_load_async_to_lds_b128(
        (gv4i_p)(Abase + (size_t)row2 * lda + k0 + colh + 8),
        (lv4i_p)(As + row2 * 40 + colh + 8), 0, 0);
    __builtin_amdgcn_global_load_async_to_lds_b128(
        (gv4i_p)(Bbase + (size_t)row2 * ldb + k0 + colh),
        (lv4i_p)(Bs + row2 * 40 + colh), 0, 0);
    __builtin_amdgcn_global_load_async_to_lds_b128(
        (gv4i_p)(Bbase + (size_t)row2 * ldb + k0 + colh + 8),
        (lv4i_p)(Bs + row2 * 40 + colh + 8), 0, 0);
    wait_async0();                    // this wave's LDS writes landed
#else
    const uint4* ga = (const uint4*)(Abase + (size_t)row2 * lda + k0 + colh);
    const uint4* gb = (const uint4*)(Bbase + (size_t)row2 * ldb + k0 + colh);
    uint4 a0 = ga[0], a1 = ga[1];
    uint4 b0 = gb[0], b1 = gb[1];
    *(uint4*)(As + row2 * 40 + colh)     = a0;
    *(uint4*)(As + row2 * 40 + colh + 8) = a1;
    *(uint4*)(Bs + row2 * 40 + colh)     = b0;
    *(uint4*)(Bs + row2 * 40 + colh + 8) = b1;
#endif
    if (k0 + 32 < Kdim) {  // speculative prefetch of next K tile -> global_prefetch_b8
      __builtin_prefetch(Abase + (size_t)row2 * lda + (k0 + 32) + colh, 0, 3);
      __builtin_prefetch(Bbase + (size_t)row2 * ldb + (k0 + 32) + colh, 0, 3);
    }
    __syncthreads();                  // all waves' tile writes visible

    // A fragment: lane m16 holds row (wave*16+m16); sel picks K 0..7/16..23 vs 8..15/24..31
    v8h alo = *(const v8h*)(As + (wave * 16 + m16) * 40 + sel * 8);
    v8h ahi = *(const v8h*)(As + (wave * 16 + m16) * 40 + 16 + sel * 8);
    v16h af = __builtin_shufflevector(alo, ahi, 0,1,2,3,4,5,6,7,8,9,10,11,12,13,14,15);
#pragma unroll
    for (int j = 0; j < 4; ++j) {
      v8h blo = *(const v8h*)(Bs + (j * 16 + m16) * 40 + sel * 8);
      v8h bhi = *(const v8h*)(Bs + (j * 16 + m16) * 40 + 16 + sel * 8);
      v16h bf = __builtin_shufflevector(blo, bhi, 0,1,2,3,4,5,6,7,8,9,10,11,12,13,14,15);
      acc[j] = __builtin_amdgcn_wmma_f32_16x16x32_f16(
          false, af, false, bf, (short)0, acc[j], false, false);
    }
  }
}

// ---------------------------------------------------------------------------
// Per-block weight prep: f32 -> f16 weight copies (padded) and A = -exp(A_log)
// segments: wA 384*96 | wX 4*64*192 (pad 38->64) | wO 128*192 (pad 96->128) | Aexp 4*192*16
// ---------------------------------------------------------------------------
__global__ void prep_kernel(const float* __restrict__ ipw, const float* __restrict__ xpw,
                            const float* __restrict__ opw, const float* __restrict__ alog,
                            _Float16* wA, _Float16* wX, _Float16* wO, float* Aexp)
{
  int idx = blockIdx.x * 256 + threadIdx.x;       // 0..122879
  if (idx < 36864) { wA[idx] = (_Float16)ipw[idx]; return; }
  idx -= 36864;
  if (idx < 49152) {
    int k  = idx / (64 * 192);
    int r  = idx - k * 64 * 192;
    int n  = r / 192;
    int dd = r - n * 192;
    float v = (n < 38) ? xpw[(k * 38 + n) * 192 + dd] : 0.f;
    wX[idx] = (_Float16)v; return;
  }
  idx -= 49152;
  if (idx < 24576) {
    int n = idx / 192, dd = idx - n * 192;
    float v = (n < 96) ? opw[n * 192 + dd] : 0.f;
    wO[idx] = (_Float16)v; return;
  }
  idx -= 24576;                                   // < 12288
  Aexp[idx] = -__expf(alog[idx]);
}

// ---------------------------------------------------------------------------
// LayerNorm over channels (96), BCHW input -> f16 row [bl][96].
// One wave per (b,l); lane covers 3 channels.
// ---------------------------------------------------------------------------
__global__ void __launch_bounds__(256) ln_kernel(const float* __restrict__ xin,
                                                 const float* __restrict__ w,
                                                 const float* __restrict__ bsh,
                                                 _Float16* __restrict__ hA)
{
  int wid  = (blockIdx.x * blockDim.x + threadIdx.x) >> 5;   // row 0..16383
  int lane = threadIdx.x & 31;
  int b = wid >> 12, l = wid & 4095;
  const float* base = xin + (size_t)b * D_MODEL * LSEQ + l;
  float v[3], s = 0.f, s2 = 0.f;
#pragma unroll
  for (int j = 0; j < 3; ++j) {
    int c = lane + j * 32;                       // 0..95
    float x = base[(size_t)c * LSEQ];
    v[j] = x; s += x; s2 += x * x;
  }
#pragma unroll
  for (int o = 16; o > 0; o >>= 1) { s += __shfl_xor(s, o, 32); s2 += __shfl_xor(s2, o, 32); }
  float mean = s * (1.f / 96.f);
  float var  = s2 * (1.f / 96.f) - mean * mean;
  float rs   = rsqrtf(var + 1e-5f);
#pragma unroll
  for (int j = 0; j < 3; ++j) {
    int c = lane + j * 32;
    hA[(size_t)wid * 96 + c] = (_Float16)((v[j] - mean) * rs * w[c] + bsh[c]);
  }
}

// in_proj GEMM: [16384 x 96] x [384 x 96]^T -> xz f32 [16384][384]
__global__ void __launch_bounds__(128) gemm_inproj_kernel(const _Float16* __restrict__ hA,
                                                          const _Float16* __restrict__ wA,
                                                          float* __restrict__ xz)
{
  __shared__ _Float16 As[64 * 40];
  __shared__ _Float16 Bs[64 * 40];
  v8f acc[4] = {};
  const int rowBase = blockIdx.x * 64, colBase = blockIdx.y * 64;
  gemm_mainloop(hA + (size_t)rowBase * 96, 96, wA + (size_t)colBase * 96, 96, 96, As, Bs, acc);
  const int lane = threadIdx.x & 31, wave = threadIdx.x >> 5;
  const int m16 = lane & 15, sel = lane >> 4;
#pragma unroll
  for (int j = 0; j < 4; ++j)
#pragma unroll
    for (int i = 0; i < 8; ++i) {
      int r = rowBase + wave * 16 + i + sel * 8;
      int c = colBase + j * 16 + m16;
      xz[(size_t)r * 384 + c] = acc[j][i];
    }
}

// Depthwise 3x3 conv + bias + SiLU; reads xz[:, :192]; writes xc f32 & f16 [bl][192]
__global__ void __launch_bounds__(256) conv_kernel(const float* __restrict__ xz,
                                                   const float* __restrict__ cw,
                                                   const float* __restrict__ cb,
                                                   float* __restrict__ xcf,
                                                   _Float16* __restrict__ xch)
{
  int idx = blockIdx.x * 256 + threadIdx.x;       // < 4*4096*192
  int e = idx % 192;
  int l = (idx / 192) & 4095;
  int b = idx / (192 * 4096);
  int h = l >> 6, w = l & 63;
  float acc = cb[e];
  const float* wp = cw + e * 9;
#pragma unroll
  for (int kh = 0; kh < 3; ++kh) {
    int hh = h + kh - 1;
    if ((unsigned)hh >= 64u) continue;
#pragma unroll
    for (int kw = 0; kw < 3; ++kw) {
      int ww2 = w + kw - 1;
      if ((unsigned)ww2 >= 64u) continue;
      acc += wp[kh * 3 + kw] * xz[((size_t)(b * LSEQ + (hh << 6) + ww2)) * 384 + e];
    }
  }
  float s = acc / (1.f + __expf(-acc));
  xcf[idx] = s;
  xch[idx] = (_Float16)s;
}

// x_proj GEMM per (b,k): [4096 x 192] x [64 x 192]^T -> dbl f32 [b*4+k][4096][38]
__global__ void __launch_bounds__(128) gemm_xproj_kernel(const _Float16* __restrict__ xch,
                                                         const _Float16* __restrict__ wX,
                                                         float* __restrict__ dblb)
{
  __shared__ _Float16 As[64 * 40];
  __shared__ _Float16 Bs[64 * 40];
  v8f acc[4] = {};
  const int z = blockIdx.z;               // b*4+k
  const int b = z >> 2, k = z & 3;
  const int rowBase = blockIdx.x * 64;
  const _Float16* A  = xch + (size_t)b * LSEQ * 192 + (size_t)rowBase * 192;
  const _Float16* Bw = wX + (size_t)k * 64 * 192;
  float* C = dblb + (size_t)z * LSEQ * 38;
  gemm_mainloop(A, 192, Bw, 192, 192, As, Bs, acc);
  const int lane = threadIdx.x & 31, wave = threadIdx.x >> 5;
  const int m16 = lane & 15, sel = lane >> 4;
#pragma unroll
  for (int j = 0; j < 4; ++j)
#pragma unroll
    for (int i = 0; i < 8; ++i) {
      int r = rowBase + wave * 16 + i + sel * 8;  // 0..4095
      int c = j * 16 + m16;
      if (c < 38) C[(size_t)r * 38 + c] = acc[j][i];
    }
}

// scan index map: direction k step t -> sequence position l in row-major space
__device__ __forceinline__ int scan_map(int k, int t)
{
  switch (k & 3) {
    case 0:  return t;
    case 1:  return ((t & 63) << 6) | (t >> 6);
    case 2:  return 4095 - t;
    default: { int tt = 4095 - t; return ((tt & 63) << 6) | (tt >> 6); }
  }
}

// Selective scan: 16-lane group per (b,k,d); lane = state n. 3072 groups.
__global__ void __launch_bounds__(256) scan_kernel(const float* __restrict__ dblb,
                                                   const float* __restrict__ xcf,
                                                   const float* __restrict__ dtw,
                                                   const float* __restrict__ dtb,
                                                   const float* __restrict__ Aexp,
                                                   const float* __restrict__ Ds,
                                                   float* __restrict__ ys4)
{
  int gid = blockIdx.x * 256 + threadIdx.x;
  int g = gid >> 4, n = gid & 15;
  int b = g / 768;
  int rem = g - b * 768;
  int k = rem / 192;
  int d = rem - k * 192;
  int kd = k * 192 + d;

  float An = Aexp[kd * 16 + n];
  float w0 = dtw[kd * 6 + 0], w1 = dtw[kd * 6 + 1], w2 = dtw[kd * 6 + 2];
  float w3 = dtw[kd * 6 + 3], w4 = dtw[kd * 6 + 4], w5 = dtw[kd * 6 + 5];
  float bias = dtb[kd], Dd = Ds[kd];

  const float* drow = dblb + (size_t)(b * 4 + k) * LSEQ * 38;
  const float* ub   = xcf + (size_t)b * LSEQ * 192 + d;
  float* yo         = ys4 + (size_t)((b * 4 + k) * 192 + d) * LSEQ;

  float h = 0.f;
  for (int t = 0; t < LSEQ; ++t) {
    int l = scan_map(k, t);
    const float* row = drow + (size_t)l * 38;
    float x = bias + row[0] * w0 + row[1] * w1 + row[2] * w2
                   + row[3] * w3 + row[4] * w4 + row[5] * w5;
    float delta = (x > 20.f) ? x : __logf(1.f + __expf(x));   // softplus
    float u  = ub[(size_t)l * 192];
    float du = delta * u;
    h = h * __expf(delta * An) + du * row[6 + n];
    float p = h * row[22 + n];
    p += __shfl_xor(p, 1, 16);
    p += __shfl_xor(p, 2, 16);
    p += __shfl_xor(p, 4, 16);
    p += __shfl_xor(p, 8, 16);
    if (n == 0) yo[l] = p + Dd * u;
  }
}

// Combine 4 directions + out-LN(192) + SiLU(z) gate -> yg f16 [bl][192]
__global__ void __launch_bounds__(256) gate_kernel(const float* __restrict__ ys4,
                                                   const float* __restrict__ xz,
                                                   const float* __restrict__ w,
                                                   const float* __restrict__ bsh,
                                                   _Float16* __restrict__ yg)
{
  int wid  = (blockIdx.x * blockDim.x + threadIdx.x) >> 5;   // row 0..16383
  int lane = threadIdx.x & 31;
  int b = wid >> 12, l = wid & 4095;
  const float* yb = ys4 + (size_t)b * 4 * 192 * LSEQ + l;
  float v[6], s = 0.f, s2 = 0.f;
#pragma unroll
  for (int j = 0; j < 6; ++j) {
    int e = lane + j * 32;                       // 0..191
    float y = yb[(size_t)e * LSEQ] + yb[(size_t)(192 + e) * LSEQ]
            + yb[(size_t)(384 + e) * LSEQ] + yb[(size_t)(576 + e) * LSEQ];
    v[j] = y; s += y; s2 += y * y;
  }
#pragma unroll
  for (int o = 16; o > 0; o >>= 1) { s += __shfl_xor(s, o, 32); s2 += __shfl_xor(s2, o, 32); }
  float mean = s * (1.f / 192.f);
  float var  = s2 * (1.f / 192.f) - mean * mean;
  float rs   = rsqrtf(var + 1e-5f);
#pragma unroll
  for (int j = 0; j < 6; ++j) {
    int e = lane + j * 32;
    float yn = (v[j] - mean) * rs * w[e] + bsh[e];
    float z  = xz[(size_t)wid * 384 + 192 + e];
    float g  = z / (1.f + __expf(-z));
    yg[(size_t)wid * 192 + e] = (_Float16)(yn * g);
  }
}

// out_proj GEMM [16384 x 192] x [128 x 192]^T, residual-add, write BCHW
__global__ void __launch_bounds__(128) gemm_outproj_kernel(const _Float16* __restrict__ yg,
                                                           const _Float16* __restrict__ wO,
                                                           const float* __restrict__ xprev,
                                                           float* __restrict__ xnext)
{
  __shared__ _Float16 As[64 * 40];
  __shared__ _Float16 Bs[64 * 40];
  v8f acc[4] = {};
  const int rowBase = blockIdx.x * 64, colBase = blockIdx.y * 64;
  gemm_mainloop(yg + (size_t)rowBase * 192, 192, wO + (size_t)colBase * 192, 192, 192, As, Bs, acc);
  const int lane = threadIdx.x & 31, wave = threadIdx.x >> 5;
  const int m16 = lane & 15, sel = lane >> 4;
#pragma unroll
  for (int j = 0; j < 4; ++j)
#pragma unroll
    for (int i = 0; i < 8; ++i) {
      int r = rowBase + wave * 16 + i + sel * 8;
      int c = colBase + j * 16 + m16;
      if (c < D_MODEL) {
        int b = r >> 12, l = r & 4095;
        size_t o = ((size_t)(b * D_MODEL + c)) * LSEQ + l;
        xnext[o] = xprev[o] + acc[j][i];
      }
    }
}

// 2x2 maxpool on BCHW
__global__ void __launch_bounds__(256) pool_kernel(const float* __restrict__ x,
                                                   float* __restrict__ out)
{
  int idx = blockIdx.x * 256 + threadIdx.x;       // < 4*96*32*32
  int ow = idx & 31, oh = (idx >> 5) & 31;
  int bc = idx >> 10;                             // b*96 + c
  const float* p = x + ((size_t)bc * 64 + oh * 2) * 64 + ow * 2;
  out[idx] = fmaxf(fmaxf(p[0], p[1]), fmaxf(p[64], p[65]));
}

// ---------------------------------------------------------------------------
extern "C" void kernel_launch(void* const* d_in, const int* in_sizes, int n_in,
                              void* d_out, int out_size, void* d_ws, size_t ws_size,
                              hipStream_t stream)
{
  const float* x_in    = (const float*)d_in[0];
  const float* ln_w    = (const float*)d_in[1];
  const float* ln_b    = (const float*)d_in[2];
  const float* in_proj = (const float*)d_in[3];
  const float* conv_w  = (const float*)d_in[4];
  const float* conv_b  = (const float*)d_in[5];
  const float* xproj_w = (const float*)d_in[6];
  const float* dt_w    = (const float*)d_in[7];
  const float* dt_b    = (const float*)d_in[8];
  const float* A_logs  = (const float*)d_in[9];
  const float* Ds      = (const float*)d_in[10];
  const float* oln_w   = (const float*)d_in[11];
  const float* oln_b   = (const float*)d_in[12];
  const float* oproj_w = (const float*)d_in[13];

  char* ws = (char*)d_ws;
  size_t off = 0;
  auto alloc = [&](size_t bytes) -> char* {
    char* p = ws + off;
    off += (bytes + 255) & ~(size_t)255;
    return p;
  };

  _Float16* hA   = (_Float16*)alloc((size_t)BL * 96 * 2);          // LN'd input, f16
  _Float16* wA   = (_Float16*)alloc((size_t)384 * 96 * 2);         // in_proj f16
  _Float16* wX   = (_Float16*)alloc((size_t)4 * 64 * 192 * 2);     // x_proj f16 padded
  _Float16* wO   = (_Float16*)alloc((size_t)128 * 192 * 2);        // out_proj f16 padded
  float*    Aexp = (float*)alloc((size_t)4 * 192 * 16 * 4);        // -exp(A_log)
  float*    xz   = (float*)alloc((size_t)BL * 384 * 4);            // in_proj out
  float*    xcf  = (float*)alloc((size_t)BL * 192 * 4);            // conv out f32
  _Float16* xch  = (_Float16*)alloc((size_t)BL * 192 * 2);         // conv out f16
  float*    dblb = (float*)alloc((size_t)16 * LSEQ * 38 * 4);      // x_proj out
  float*    ys4  = (float*)alloc((size_t)16 * 192 * LSEQ * 4);     // scan out per dir
  _Float16* yg   = (_Float16*)alloc((size_t)BL * 192 * 2);         // gated y f16
  float*    xtmp = (float*)alloc((size_t)BATCH * D_MODEL * LSEQ * 4); // block-0 output

  float* xfinal   = (float*)d_out;
  float* pool_out = xfinal + (size_t)BATCH * D_MODEL * LSEQ;

  for (int i = 0; i < 2; ++i) {
    const float* xin = (i == 0) ? x_in : xtmp;
    float* xout      = (i == 0) ? xtmp : xfinal;

    prep_kernel<<<480, 256, 0, stream>>>(
        in_proj + (size_t)i * 384 * 96, xproj_w + (size_t)i * 4 * 38 * 192,
        oproj_w + (size_t)i * 96 * 192, A_logs + (size_t)i * 4 * 192 * 16,
        wA, wX, wO, Aexp);

    ln_kernel<<<2048, 256, 0, stream>>>(xin, ln_w + i * 96, ln_b + i * 96, hA);

    gemm_inproj_kernel<<<dim3(256, 6, 1), 128, 0, stream>>>(hA, wA, xz);

    conv_kernel<<<12288, 256, 0, stream>>>(
        xz, conv_w + (size_t)i * 192 * 9, conv_b + (size_t)i * 192, xcf, xch);

    gemm_xproj_kernel<<<dim3(64, 1, 16), 128, 0, stream>>>(xch, wX, dblb);

    scan_kernel<<<192, 256, 0, stream>>>(
        dblb, xcf, dt_w + (size_t)i * 4 * 192 * 6, dt_b + (size_t)i * 4 * 192,
        Aexp, Ds + (size_t)i * 4 * 192, ys4);

    gate_kernel<<<2048, 256, 0, stream>>>(ys4, xz, oln_w + i * 192, oln_b + i * 192, yg);

    gemm_outproj_kernel<<<dim3(256, 2, 1), 128, 0, stream>>>(yg, wO, xin, xout);
  }

  pool_kernel<<<1536, 256, 0, stream>>>(xfinal, pool_out);
}
